// QQwen2DecoderLayer_46231027974640
// MI455X (gfx1250) — compile-verified, matrix operand order checked by reading, not run on hardware
//
#include <hip/hip_runtime.h>
#include <hip/hip_bf16.h>
#include <stdint.h>

// ---------------- problem constants ----------------
#define B_     2
#define S_     2048
#define H_     1536
#define NH_    12
#define NKV_   2
#define HD_    128
#define INTER_ 8960
#define BS_    (B_ * S_)
#define NREP_  (NH_ / NKV_)
#define LOG_THETA 13.815510557964274f   // ln(1e6)

typedef __attribute__((ext_vector_type(16))) _Float16 v16h;
typedef __attribute__((ext_vector_type(8)))  float    v8f;

// ---------------- WMMA helpers ----------------
__device__ __forceinline__ v16h load_frag16(const _Float16* p0, const _Float16* p1) {
    union { v16h v; uint4 u[2]; } f;
    f.u[0] = *reinterpret_cast<const uint4*>(p0);
    f.u[1] = *reinterpret_cast<const uint4*>(p1);
    return f.v;
}

__device__ __forceinline__ v8f wmma16(v16h a, v16h b, v8f c) {
    // D(16x16 f32) = A(16x32 f16) * B(32x16 f16) + C
    return __builtin_amdgcn_wmma_f32_16x16x32_f16(false, a, false, b, (short)0, c,
                                                  false, false);
}

// ---------------- elementwise / prep kernels ----------------

// per-row RMSNorm: y = x * rsqrt(mean(x^2)+eps) * w   (one block per row, K elems)
__global__ __launch_bounds__(256)
void rmsnorm_kernel(const float* __restrict__ x, const float* __restrict__ w,
                    float* __restrict__ y, int K) {
    __shared__ float red[256];
    const size_t row = blockIdx.x;
    const float* xr = x + row * (size_t)K;
    float* yr = y + row * (size_t)K;
    float ss = 0.f;
    for (int j = threadIdx.x; j < K; j += 256) { float v = xr[j]; ss += v * v; }
    red[threadIdx.x] = ss; __syncthreads();
    for (int s = 128; s > 0; s >>= 1) {
        if ((int)threadIdx.x < s) red[threadIdx.x] += red[threadIdx.x + s];
        __syncthreads();
    }
    const float rs = rsqrtf(red[0] / (float)K + 1e-6f);
    for (int j = threadIdx.x; j < K; j += 256) yr[j] = xr[j] * rs * w[j];
}

// permute + segmentwise fake-quant (8/6/4 bit) -> f16, one block per row
__global__ __launch_bounds__(256)
void quant_perm_kernel(const float* __restrict__ x, const int* __restrict__ perm,
                       _Float16* __restrict__ out, int K, int p8, int p6) {
    extern __shared__ float rowbuf[];
    __shared__ float red[256];
    __shared__ float segscale[3];
    const size_t row = blockIdx.x;
    const float* xr = x + row * (size_t)K;
    for (int j = threadIdx.x; j < K; j += 256) rowbuf[j] = xr[perm[j]];
    __syncthreads();
    const int seg_lo[3] = {0, p8, p8 + p6};
    const int seg_hi[3] = {p8, p8 + p6, K};
    const float qmaxs[3] = {127.f, 31.f, 7.f};
    for (int sg = 0; sg < 3; ++sg) {
        float mx = 0.f;
        for (int j = seg_lo[sg] + (int)threadIdx.x; j < seg_hi[sg]; j += 256)
            mx = fmaxf(mx, fabsf(rowbuf[j]));
        red[threadIdx.x] = mx; __syncthreads();
        for (int s = 128; s > 0; s >>= 1) {
            if ((int)threadIdx.x < s) red[threadIdx.x] = fmaxf(red[threadIdx.x], red[threadIdx.x + s]);
            __syncthreads();
        }
        if (threadIdx.x == 0) segscale[sg] = fmaxf(red[0] / qmaxs[sg], 1e-8f);
        __syncthreads();
    }
    _Float16* orow = out + row * (size_t)K;
    for (int j = threadIdx.x; j < K; j += 256) {
        int sg = (j < p8) ? 0 : ((j < p8 + p6) ? 1 : 2);
        float s = segscale[sg], qm = qmaxs[sg];
        float q = fminf(fmaxf(rintf(rowbuf[j] / s), -qm - 1.f), qm) * s;
        orow[j] = (_Float16)q;
    }
}

// weight column-permute + cast to f16: out[n,j] = (f16) w[n, perm[j]]
__global__ __launch_bounds__(256)
void permw_f16_kernel(const float* __restrict__ w, const int* __restrict__ perm,
                      _Float16* __restrict__ out, long total, int K) {
    long idx = (long)blockIdx.x * 256 + threadIdx.x;
    if (idx >= total) return;
    long n = idx / K; int j = (int)(idx % K);
    out[idx] = (_Float16)w[n * (long)K + perm[j]];
}

// RoPE on Q, layout -> [B, NH, S, HD] f16
__global__ __launch_bounds__(256)
void rope_q_kernel(const float* __restrict__ q, const int* __restrict__ pos_ids,
                   _Float16* __restrict__ qh) {
    long total = (long)BS_ * NH_ * HD_;
    long idx = (long)blockIdx.x * 256 + threadIdx.x;
    if (idx >= total) return;
    int d = (int)(idx & (HD_ - 1));
    long t = idx >> 7;                // (b*s, h)
    int hh = (int)(t % NH_);
    long bs = t / NH_;
    int s = (int)(bs % S_);
    int b = (int)(bs / S_);
    const float* row = q + bs * (long)H_ + hh * HD_;
    float v = row[d];
    float other = (d < 64) ? -row[d + 64] : row[d - 64];
    float pos = (float)pos_ids[bs];
    float ang = pos * __expf(-(float)(d & 63) * (LOG_THETA / 64.0f));
    float o = v * cosf(ang) + other * sinf(ang);
    qh[(((long)b * NH_ + hh) * S_ + s) * HD_ + d] = (_Float16)o;
}

// K: int4 group fake-quant (group=HD) then RoPE, layout -> [B, NKV, S, HD] f16
__global__ __launch_bounds__(128)
void k_quant_rope_kernel(const float* __restrict__ k, const int* __restrict__ pos_ids,
                         _Float16* __restrict__ kh) {
    __shared__ float buf[HD_];
    __shared__ float red[HD_];
    long g = blockIdx.x;                 // b*s*NKV blocks
    int kvh = (int)(g % NKV_);
    long bs = g / NKV_;
    int s = (int)(bs % S_);
    int b = (int)(bs / S_);
    int d = threadIdx.x;
    float v = k[bs * (long)(NKV_ * HD_) + kvh * HD_ + d];
    red[d] = fabsf(v); __syncthreads();
    for (int st = 64; st > 0; st >>= 1) {
        if (d < st) red[d] = fmaxf(red[d], red[d + st]);
        __syncthreads();
    }
    float sc = fmaxf(red[0] / 7.f, 1e-8f);
    float q = fminf(fmaxf(rintf(v / sc), -8.f), 7.f) * sc;
    buf[d] = q; __syncthreads();
    float other = (d < 64) ? -buf[d + 64] : buf[d - 64];
    float pos = (float)pos_ids[bs];
    float ang = pos * __expf(-(float)(d & 63) * (LOG_THETA / 64.0f));
    float o = q * cosf(ang) + other * sinf(ang);
    kh[(((long)b * NKV_ + kvh) * S_ + s) * HD_ + d] = (_Float16)o;
}

// V: int4 group fake-quant, transposed layout -> [B, NKV, HD, S] f16
__global__ __launch_bounds__(128)
void v_quant_t_kernel(const float* __restrict__ v, _Float16* __restrict__ vt) {
    __shared__ float red[HD_];
    long g = blockIdx.x;
    int kvh = (int)(g % NKV_);
    long bs = g / NKV_;
    int s = (int)(bs % S_);
    int b = (int)(bs / S_);
    int d = threadIdx.x;
    float x = v[bs * (long)(NKV_ * HD_) + kvh * HD_ + d];
    red[d] = fabsf(x); __syncthreads();
    for (int st = 64; st > 0; st >>= 1) {
        if (d < st) red[d] = fmaxf(red[d], red[d + st]);
        __syncthreads();
    }
    float sc = fmaxf(red[0] / 7.f, 1e-8f);
    float q = fminf(fmaxf(rintf(x / sc), -8.f), 7.f) * sc;
    vt[(((long)b * NKV_ + kvh) * HD_ + d) * S_ + s] = (_Float16)q;
}

// residual add: out = a + b
__global__ __launch_bounds__(256)
void add_kernel(const float* __restrict__ a, const float* __restrict__ b,
                float* __restrict__ out, long n) {
    long i = (long)blockIdx.x * 256 + threadIdx.x;
    if (i < n) out[i] = a[i] + b[i];
}

// ---------------- WMMA GEMM-NT: C[M,N] = A[M,K] * W[N,K]^T (+bias / fused silu-mul)
// block: 128 threads (4 waves), block tile 256(M) x 64(N), wave tile 64x64, k-step 32
// per k-step: 8 fragment loads (16x b128) -> 16 WMMAs  (1:1 load:wmma ratio)
__global__ __launch_bounds__(128)
void gemm_nt_wmma_kernel(const _Float16* __restrict__ A, const _Float16* __restrict__ W,
                         const float* __restrict__ bias, float* __restrict__ C,
                         float* __restrict__ fuse, int M, int N, int K, int epi) {
    const int lane = threadIdx.x & 31;
    const int wave = threadIdx.x >> 5;
    const int l15  = lane & 15;
    const int hi   = lane >> 4;
    const int boff = hi * 8;            // 16-bit A/B operand: chunks at k0+hi*8, k0+16+hi*8
    const int n0 = blockIdx.x * 64;
    const int m0 = blockIdx.y * 256 + wave * 64;

    const v8f vzero = {};
    v8f acc[4][4];
#pragma unroll
    for (int i = 0; i < 4; ++i)
#pragma unroll
        for (int j = 0; j < 4; ++j) acc[i][j] = vzero;

    const _Float16* ap[4];
#pragma unroll
    for (int mt = 0; mt < 4; ++mt) ap[mt] = A + (size_t)(m0 + 16 * mt + l15) * K;
    const _Float16* bp[4];
#pragma unroll
    for (int t = 0; t < 4; ++t) bp[t] = W + (size_t)(n0 + 16 * t + l15) * K;

    for (int k = 0; k < K; k += 32) {
        if (k + 128 < K) {
            __builtin_prefetch(ap[0] + k + 128, 0, 3);   // global_prefetch_b8
            __builtin_prefetch(bp[0] + k + 128, 0, 3);
        }
        v16h afr[4];
#pragma unroll
        for (int mt = 0; mt < 4; ++mt)
            afr[mt] = load_frag16(ap[mt] + k + boff, ap[mt] + k + 16 + boff);
#pragma unroll
        for (int t = 0; t < 4; ++t) {
            v16h bfr = load_frag16(bp[t] + k + boff, bp[t] + k + 16 + boff);
#pragma unroll
            for (int mt = 0; mt < 4; ++mt)
                acc[mt][t] = wmma16(afr[mt], bfr, acc[mt][t]);
        }
    }

#pragma unroll
    for (int mt = 0; mt < 4; ++mt) {
#pragma unroll
        for (int t = 0; t < 4; ++t) {
#pragma unroll
            for (int r = 0; r < 8; ++r) {
                int row = m0 + mt * 16 + r + hi * 8;   // C layout: VGPR r -> M=r / r+8
                int col = n0 + t * 16 + l15;
                float v = acc[mt][t][r];
                if (bias) v += bias[col];
                size_t idx = (size_t)row * N + col;
                if (epi == 1) {
                    float gv = fuse[idx];                     // h = silu(g) * u
                    fuse[idx] = (gv / (1.0f + __expf(-gv))) * v;
                } else {
                    C[idx] = v;
                }
            }
        }
    }
}

// ---------------- flash attention (causal, GQA) ----------------
// grid: (S/128, NH, B), block 256 (8 waves). Each wave owns 16 q-rows.
__global__ __launch_bounds__(256)
void flash_attn_kernel(const _Float16* __restrict__ qh, const _Float16* __restrict__ kh,
                       const _Float16* __restrict__ vt, float* __restrict__ attn_out) {
    const int lane = threadIdx.x & 31;
    const int wave = threadIdx.x >> 5;
    const int l15  = lane & 15;
    const int hi   = lane >> 4;
    const int boff = hi * 8;
    const int q0 = blockIdx.x * 128;
    const int h  = blockIdx.y;
    const int b  = blockIdx.z;
    const int kvh = h / NREP_;

    const _Float16* Q  = qh + ((size_t)b * NH_  + h)   * (size_t)S_ * HD_;
    const _Float16* Kp = kh + ((size_t)b * NKV_ + kvh) * (size_t)S_ * HD_;
    const _Float16* Vt = vt + ((size_t)b * NKV_ + kvh) * (size_t)HD_ * S_;

    __shared__ __align__(16) _Float16 ldsP[8][16][128];   // per-wave P tiles (32 KB)

    // preload Q A-fragments (HD=128 -> 4 k-steps)
    const _Float16* Qrow = Q + (size_t)(q0 + wave * 16 + l15) * HD_;
    v16h qa[4];
#pragma unroll
    for (int kk = 0; kk < 4; ++kk)
        qa[kk] = load_frag16(Qrow + kk * 32 + boff, Qrow + kk * 32 + 16 + boff);

    const v8f vzero = {};
    v8f o[8];
#pragma unroll
    for (int dt = 0; dt < 8; ++dt) o[dt] = vzero;
    float mrun[8], lrun[8];
#pragma unroll
    for (int r = 0; r < 8; ++r) { mrun[r] = -1e30f; lrun[r] = 0.f; }

    const float sc = 0.088388347648318447f;   // 1/sqrt(128)

    for (int j0 = 0; j0 <= q0; j0 += 128) {
        // ---- S = Q * K^T ----
        v8f s[8];
#pragma unroll
        for (int t = 0; t < 8; ++t) s[t] = vzero;
#pragma unroll
        for (int kk = 0; kk < 4; ++kk) {
#pragma unroll
            for (int t = 0; t < 8; ++t) {
                const _Float16* krow = Kp + (size_t)(j0 + 16 * t + l15) * HD_ + kk * 32 + boff;
                v16h kb = load_frag16(krow, krow + 16);
                s[t] = wmma16(qa[kk], kb, s[t]);
            }
        }
        // ---- scale + causal mask ----
#pragma unroll
        for (int t = 0; t < 8; ++t) {
#pragma unroll
            for (int r = 0; r < 8; ++r) {
                float v = s[t][r] * sc;
                int row_g = q0 + wave * 16 + r + hi * 8;
                int col   = j0 + 16 * t + l15;
                if (col > row_g) v = -1e30f;
                s[t][r] = v;
            }
        }
        // ---- online softmax (per row: reduce over 8 tiles x 16 lanes of half-wave) ----
#pragma unroll
        for (int r = 0; r < 8; ++r) {
            float mx = -1e30f;
#pragma unroll
            for (int t = 0; t < 8; ++t) mx = fmaxf(mx, s[t][r]);
            mx = fmaxf(mx, __shfl_xor(mx, 1, 32));
            mx = fmaxf(mx, __shfl_xor(mx, 2, 32));
            mx = fmaxf(mx, __shfl_xor(mx, 4, 32));
            mx = fmaxf(mx, __shfl_xor(mx, 8, 32));
            float mn = fmaxf(mrun[r], mx);
            float scale = __expf(mrun[r] - mn);
            float sum = 0.f;
#pragma unroll
            for (int t = 0; t < 8; ++t) {
                float p = __expf(s[t][r] - mn);
                s[t][r] = p;
                sum += p;
            }
            sum += __shfl_xor(sum, 1, 32);
            sum += __shfl_xor(sum, 2, 32);
            sum += __shfl_xor(sum, 4, 32);
            sum += __shfl_xor(sum, 8, 32);
            lrun[r] = lrun[r] * scale + sum;
            mrun[r] = mn;
#pragma unroll
            for (int dt = 0; dt < 8; ++dt) o[dt][r] *= scale;
        }
        // ---- P -> LDS (f16), re-shape accumulator layout to A-operand layout ----
#pragma unroll
        for (int t = 0; t < 8; ++t)
#pragma unroll
            for (int r = 0; r < 8; ++r)
                ldsP[wave][r + hi * 8][t * 16 + l15] = (_Float16)s[t][r];
        // same-wave DS ops are in order; reads below see the writes above
        // ---- O += P * V^T (Vt is [HD, S]) ----
        const _Float16* prow = &ldsP[wave][l15][0];
#pragma unroll
        for (int kk = 0; kk < 4; ++kk) {
            v16h pa = load_frag16(prow + kk * 32 + boff, prow + kk * 32 + 16 + boff);
#pragma unroll
            for (int dt = 0; dt < 8; ++dt) {
                const _Float16* vrow = Vt + (size_t)(dt * 16 + l15) * S_ + j0 + kk * 32 + boff;
                v16h vb = load_frag16(vrow, vrow + 16);
                o[dt] = wmma16(pa, vb, o[dt]);
            }
        }
    }

    // ---- write out: [B*S, NH*HD] f32 ----
#pragma unroll
    for (int dt = 0; dt < 8; ++dt) {
#pragma unroll
        for (int r = 0; r < 8; ++r) {
            int row_g = q0 + wave * 16 + r + hi * 8;
            int col   = h * HD_ + dt * 16 + l15;
            attn_out[((size_t)b * S_ + row_g) * (size_t)H_ + col] = o[dt][r] / lrun[r];
        }
    }
}

// ---------------- host-side launch ----------------
static inline size_t align256(size_t x) { return (x + 255) & ~(size_t)255; }

extern "C" void kernel_launch(void* const* d_in, const int* in_sizes, int n_in,
                              void* d_out, int out_size, void* d_ws, size_t ws_size,
                              hipStream_t stream) {
    const float* hidden  = (const float*)d_in[0];
    const int*   pos_ids = (const int*)  d_in[1];
    const float* q_w  = (const float*)d_in[2];
    const float* q_b  = (const float*)d_in[3];
    const float* k_w  = (const float*)d_in[4];
    const float* k_b  = (const float*)d_in[5];
    const float* v_w  = (const float*)d_in[6];
    const float* v_b  = (const float*)d_in[7];
    const float* o_w  = (const float*)d_in[8];
    const float* gate_w = (const float*)d_in[9];
    const float* up_w   = (const float*)d_in[10];
    const float* down_w = (const float*)d_in[11];
    const float* ln1_w  = (const float*)d_in[12];
    const float* ln2_w  = (const float*)d_in[13];
    const int* perm_qkv  = (const int*)d_in[14];
    const int* perm_o    = (const int*)d_in[15];
    const int* perm_gate = (const int*)d_in[16];
    const int* perm_down = (const int*)d_in[17];
    float* out = (float*)d_out;

    // workspace carve-up
    char* ws = (char*)d_ws;
    size_t off = 0;
    auto alloc = [&](size_t bytes) -> void* {
        void* p = ws + off;
        off = align256(off + bytes);
        return p;
    };
    _Float16* wpq = (_Float16*)alloc((size_t)H_ * H_ * 2);
    _Float16* wpk = (_Float16*)alloc((size_t)NKV_ * HD_ * H_ * 2);
    _Float16* wpv = (_Float16*)alloc((size_t)NKV_ * HD_ * H_ * 2);
    _Float16* wpo = (_Float16*)alloc((size_t)H_ * H_ * 2);
    _Float16* wpg = (_Float16*)alloc((size_t)INTER_ * H_ * 2);
    _Float16* wpu = (_Float16*)alloc((size_t)INTER_ * H_ * 2);
    _Float16* wpd = (_Float16*)alloc((size_t)H_ * INTER_ * 2);
    float*    xn   = (float*)   alloc((size_t)BS_ * H_ * 4);
    _Float16* xq   = (_Float16*)alloc((size_t)BS_ * H_ * 2);
    float*    qbuf = (float*)   alloc((size_t)BS_ * H_ * 4);
    float*    kbuf = (float*)   alloc((size_t)BS_ * NKV_ * HD_ * 4);
    float*    vbuf = (float*)   alloc((size_t)BS_ * NKV_ * HD_ * 4);
    _Float16* qh   = (_Float16*)alloc((size_t)BS_ * NH_ * HD_ * 2);
    _Float16* kh   = (_Float16*)alloc((size_t)BS_ * NKV_ * HD_ * 2);
    _Float16* vt   = (_Float16*)alloc((size_t)BS_ * NKV_ * HD_ * 2);
    float*    attn = (float*)   alloc((size_t)BS_ * H_ * 4);
    float*    oprj = (float*)   alloc((size_t)BS_ * H_ * 4);
    float*    gbuf = (float*)   alloc((size_t)BS_ * INTER_ * 4);
    _Float16* hq   = (_Float16*)alloc((size_t)BS_ * INTER_ * 2);
    float*    dprj = (float*)   alloc((size_t)BS_ * H_ * 4);
    (void)ws_size; (void)n_in; (void)in_sizes; (void)out_size;

    // 1) permute + cast weights to f16
    auto permw = [&](const float* w, const int* perm, _Float16* dst, long N, int K) {
        long total = N * (long)K;
        permw_f16_kernel<<<(int)((total + 255) / 256), 256, 0, stream>>>(w, perm, dst, total, K);
    };
    permw(q_w,    perm_qkv,  wpq, H_,          H_);
    permw(k_w,    perm_qkv,  wpk, NKV_ * HD_,  H_);
    permw(v_w,    perm_qkv,  wpv, NKV_ * HD_,  H_);
    permw(o_w,    perm_o,    wpo, H_,          H_);
    permw(gate_w, perm_gate, wpg, INTER_,      H_);
    permw(up_w,   perm_gate, wpu, INTER_,      H_);
    permw(down_w, perm_down, wpd, H_,          INTER_);

    // 2) rmsnorm1 + activation quant (perm_qkv, 128/1344/64)
    rmsnorm_kernel<<<BS_, 256, 0, stream>>>(hidden, ln1_w, xn, H_);
    quant_perm_kernel<<<BS_, 256, H_ * sizeof(float), stream>>>(xn, perm_qkv, xq, H_, 128, 1344);

    // 3) Q/K/V GEMMs (+bias)
    auto gemm = [&](const _Float16* A, const _Float16* W, const float* bias, float* C,
                    float* fuse, int M, int N, int K, int epi) {
        dim3 grid(N / 64, M / 256);
        gemm_nt_wmma_kernel<<<grid, 128, 0, stream>>>(A, W, bias, C, fuse, M, N, K, epi);
    };
    gemm(xq, wpq, q_b, qbuf, nullptr, BS_, H_,         H_, 0);
    gemm(xq, wpk, k_b, kbuf, nullptr, BS_, NKV_ * HD_, H_, 0);
    gemm(xq, wpv, v_b, vbuf, nullptr, BS_, NKV_ * HD_, H_, 0);

    // 4) RoPE(Q); int4-quant(K)+RoPE; int4-quant(V)+transpose
    {
        long total = (long)BS_ * NH_ * HD_;
        rope_q_kernel<<<(int)((total + 255) / 256), 256, 0, stream>>>(qbuf, pos_ids, qh);
    }
    k_quant_rope_kernel<<<BS_ * NKV_, HD_, 0, stream>>>(kbuf, pos_ids, kh);
    v_quant_t_kernel<<<BS_ * NKV_, HD_, 0, stream>>>(vbuf, vt);

    // 5) causal GQA flash attention
    {
        dim3 grid(S_ / 128, NH_, B_);
        flash_attn_kernel<<<grid, 256, 0, stream>>>(qh, kh, vt, attn);
    }

    // 6) O-projection (quantized activations, perm_o) + residual
    quant_perm_kernel<<<BS_, 256, H_ * sizeof(float), stream>>>(attn, perm_o, xq, H_, 128, 1344);
    gemm(xq, wpo, nullptr, oprj, nullptr, BS_, H_, H_, 0);
    {
        long n = (long)BS_ * H_;
        add_kernel<<<(int)((n + 255) / 256), 256, 0, stream>>>(hidden, oprj, out, n);
    }

    // 7) MLP: rmsnorm2 -> quant -> gate ; up fused with silu-mul -> h in gbuf
    rmsnorm_kernel<<<BS_, 256, 0, stream>>>(out, ln2_w, xn, H_);
    quant_perm_kernel<<<BS_, 256, H_ * sizeof(float), stream>>>(xn, perm_gate, xq, H_, 128, 1344);
    gemm(xq, wpg, nullptr, gbuf, nullptr, BS_, INTER_, H_, 0);
    gemm(xq, wpu, nullptr, nullptr, gbuf, BS_, INTER_, H_, 1);   // gbuf = silu(gbuf) * up

    // 8) down projection (quant over INTER, 512/8192/256) + final residual
    quant_perm_kernel<<<BS_, 256, INTER_ * sizeof(float), stream>>>(gbuf, perm_down, hq, INTER_, 512, 8192);
    gemm(hq, wpd, nullptr, dprj, nullptr, BS_, H_, INTER_, 0);
    {
        long n = (long)BS_ * H_;
        add_kernel<<<(int)((n + 255) / 256), 256, 0, stream>>>(out, dprj, out, n);
    }
}
